// IGNN_74217034875029
// MI455X (gfx1250) — matrix operand order repaired and checked by reading, not compiled, not run on hardware
//
#include <hip/hip_runtime.h>
#include <hip/hip_bf16.h>

typedef __attribute__((ext_vector_type(16))) _Float16 v16h;
typedef __attribute__((ext_vector_type(8)))  float    v8f;
typedef _Float16 f16;

#define NHID   64
#define NFEATC 128
#define NCLASS 10
#define NGRAPH 128
#define KAPPA  0.9f
#define NITER  25

union H2U { f16 h[2]; unsigned int u; };

// ---------------------------------------------------------------------------
// Transpose (F x N) f32 row-major -> (N x F) f16 row-major, LDS-tiled.
// ---------------------------------------------------------------------------
__global__ void k_transpose_h(const float* __restrict__ in, f16* __restrict__ out,
                              int N, int F) {
    __shared__ float tile[32][33];
    int bx = blockIdx.x * 32;  // node base
    int by = blockIdx.y * 32;  // feat base
    for (int ty = threadIdx.y; ty < 32; ty += blockDim.y) {
        int f = by + ty, n = bx + threadIdx.x;
        tile[ty][threadIdx.x] = (f < F && n < N) ? in[(size_t)f * N + n] : 0.f;
    }
    __syncthreads();
    for (int ty = threadIdx.y; ty < 32; ty += blockDim.y) {
        int n = bx + ty, f = by + threadIdx.x;
        if (n < N && f < F) out[(size_t)n * F + f] = (f16)tile[threadIdx.x][ty];
    }
}

// ---------------------------------------------------------------------------
// f32 -> f16 convert (for Omega matrices).
// ---------------------------------------------------------------------------
__global__ void k_cvt_h(const float* __restrict__ in, f16* __restrict__ out, int n) {
    int i = blockIdx.x * blockDim.x + threadIdx.x;
    if (i < n) out[i] = (f16)in[i];
}

// ---------------------------------------------------------------------------
// projection_norm_inf: project each row of W (KxK) onto L1 ball of radius v.
// One thread per row (K=64); emits f16 result for the WMMA pipeline.
// ---------------------------------------------------------------------------
__global__ void k_proj(const float* __restrict__ W, f16* __restrict__ Wp,
                       int K, float v) {
    int r = threadIdx.x;
    if (r >= K) return;
    float a[64], s[64];
    float asum = 0.f;
    for (int i = 0; i < K; ++i) {
        float w = W[r * K + i];
        a[i] = fabsf(w);
        s[i] = a[i];
        asum += a[i];
    }
    for (int i = 0; i < K - 1; ++i) {           // selection sort descending
        int mx = i;
        for (int j = i + 1; j < K; ++j) if (s[j] > s[mx]) mx = j;
        float t = s[i]; s[i] = s[mx]; s[mx] = t;
    }
    float csum = 0.f;
    int cnt = 0;
    for (int k = 1; k <= K; ++k) {
        csum += s[k - 1];
        if (s[k - 1] * (float)k > csum - v) cnt++;
    }
    int rho = cnt > 0 ? cnt : 1;
    csum = 0.f;
    for (int k = 0; k < rho; ++k) csum += s[k];
    float theta = (csum - v) / (float)rho;
    for (int i = 0; i < K; ++i) {
        float w = W[r * K + i];
        float sign = (w > 0.f) ? 1.f : ((w < 0.f) ? -1.f : 0.f);
        float pr = sign * fmaxf(a[i] - theta, 0.f);
        Wp[r * K + i] = (f16)((asum > v) ? pr : w);
    }
}

// ---------------------------------------------------------------------------
// Zero-fill (32-bit words).
// ---------------------------------------------------------------------------
__global__ void k_zero(unsigned int* __restrict__ p, size_t nwords) {
    size_t i = (size_t)blockIdx.x * blockDim.x + threadIdx.x;
    size_t stride = (size_t)gridDim.x * blockDim.x;
    for (; i < nwords; i += stride) p[i] = 0u;
}

// ---------------------------------------------------------------------------
// SpMM scatter in f16:  Yh[dst[e]][:] += w[e] * Xh[src[e]][:]
// Node vector = 64 f16 = 128B; wave32 -> one f16x2 per lane, one packed-f16
// atomic (global_atomic_pk_add_f16) per lane instead of two f32 atomics.
// Edge scalars (w/src/dst) are wave-uniform (all lanes share e = i>>5).
// ---------------------------------------------------------------------------
__global__ void k_spmm_h(const f16* __restrict__ Xh, const int* __restrict__ src,
                         const int* __restrict__ dst, const float* __restrict__ w,
                         f16* __restrict__ Yh, int E) {
    size_t total = (size_t)E * 32;
    size_t i = (size_t)blockIdx.x * blockDim.x + threadIdx.x;
    size_t stride = (size_t)gridDim.x * blockDim.x;
    for (; i < total; i += stride) {
        int e = (int)(i >> 5);
        int l = (int)(i & 31);
        float wv = w[e];
        int s = src[e], d = dst[e];
        H2U x;
        x.u = *(const unsigned int*)(Xh + (size_t)s * 64 + 2 * l);
        H2U m;
        m.h[0] = (f16)(wv * (float)x.h[0]);
        m.h[1] = (f16)(wv * (float)x.h[1]);
        f16* yp = Yh + (size_t)d * 64 + 2 * l;
        asm volatile("global_atomic_pk_add_f16 %0, %1, off"
                     :: "v"(yp), "v"(m.u) : "memory");
    }
}

// ---------------------------------------------------------------------------
// WMMA GEMM:  out[N x 64](f16) = act( A[N x K](f16) * B^T + bias ),
// B = Bmat (64 x K, f16).  out[node][n] = sum_k A[node][k] * Bmat[n][k].
// Bmat staged into LDS via async-to-LDS DMA, then ALL B fragments hoisted
// into VGPRs once per wave (B is constant across node tiles). Steady-state
// inner loop: batched b128 A loads -> one wait -> back-to-back
// v_wmma_f32_16x16x32_f16 on independent accumulators, no LDS traffic.
// ---------------------------------------------------------------------------
template <int K, bool HAS_BIAS, bool RELU>
__global__ void __launch_bounds__(256)
k_gemm_wmma(const f16* __restrict__ A,
            const f16* __restrict__ Bmat,      // 64 x K row-major f16
            const f16* __restrict__ bias,      // N x 64 node-major f16 (or unused)
            f16* __restrict__ out, int N) {
    constexpr int KF = K / 32;                 // K fragments per tile
    __shared__ f16 sB[64 * K];                 // only LDS object -> offset 0

    // async DMA: 16B chunks of Bmat into LDS, then drain ASYNCcnt + barrier
    constexpr int nchunks = (64 * K) / 8;      // 8 f16 per 16B chunk
    for (int i = threadIdx.x; i < nchunks; i += blockDim.x) {
        unsigned int lds_off = (unsigned int)(i * 16);
        const f16* gptr = Bmat + (size_t)i * 8;
        asm volatile("global_load_async_to_lds_b128 %0, %1, off"
                     :: "v"(lds_off), "v"(gptr) : "memory");
    }
    asm volatile("s_wait_asynccnt 0x0" ::: "memory");
    __syncthreads();

    int lane = threadIdx.x & 31;
    int m = lane & 15;                 // A row within tile / B column within tile
    int aoff = (lane < 16) ? 0 : 8;    // 16-bit A 16x32 layout K sub-offset
    int boff = (lane < 16) ? 0 : 8;    // 16-bit B 32x16 layout K sub-offset (matches A)

    // hoist all B fragments into registers (shared by every node tile)
    v16h bf[4][KF];
#pragma unroll
    for (int t = 0; t < 4; ++t)
#pragma unroll
        for (int kf = 0; kf < KF; ++kf) {
            const f16* brow = sB + (t * 16 + m) * K + kf * 32 + boff;
#pragma unroll
            for (int i = 0; i < 16; ++i) bf[t][kf][i] = brow[(i < 8) ? i : i + 8];
        }

    int wid = (int)((blockIdx.x * blockDim.x + threadIdx.x) >> 5);
    int nwaves = (int)((gridDim.x * blockDim.x) >> 5);
    int ntiles = N >> 4;

    for (int tile = wid; tile < ntiles; tile += nwaves) {
        // batch-load all A fragments for this 16-node strip
        const f16* Arow = A + (size_t)(tile * 16 + m) * K;
        v16h af[KF];
#pragma unroll
        for (int kf = 0; kf < KF; ++kf) {
            const f16* p0 = Arow + kf * 32 + aoff;
            const f16* p1 = Arow + kf * 32 + 16 + aoff;
#pragma unroll
            for (int i = 0; i < 8; ++i) af[kf][i] = p0[i];
#pragma unroll
            for (int i = 0; i < 8; ++i) af[kf][8 + i] = p1[i];
        }

        v8f acc[4];
#pragma unroll
        for (int t = 0; t < 4; ++t)
#pragma unroll
            for (int r = 0; r < 8; ++r) acc[t][r] = 0.f;

#pragma unroll
        for (int kf = 0; kf < KF; ++kf)
#pragma unroll
            for (int t = 0; t < 4; ++t)
                acc[t] = __builtin_amdgcn_wmma_f32_16x16x32_f16(
                    false, af[kf], false, bf[t][kf], (short)0, acc[t],
                    false, false);

        int nodebase = tile * 16 + ((lane < 16) ? 0 : 8);
#pragma unroll
        for (int t = 0; t < 4; ++t) {
            int n = t * 16 + m;
#pragma unroll
            for (int r = 0; r < 8; ++r) {
                size_t idx = (size_t)(nodebase + r) * 64 + n;
                float vv = acc[t][r];
                if (HAS_BIAS) vv += (float)bias[idx];
                if (RELU) vv = fmaxf(vv, 0.f);
                out[idx] = (f16)vv;
            }
        }
    }
}

// ---------------------------------------------------------------------------
// Global add pool:  g[batch[n]][h] += x[n][h]   (f16 in, f32 accumulate)
// ---------------------------------------------------------------------------
__global__ void k_pool(const f16* __restrict__ Xh, const int* __restrict__ batch,
                       float* __restrict__ g, int N) {
    size_t total = (size_t)N * 64;
    size_t i = (size_t)blockIdx.x * blockDim.x + threadIdx.x;
    size_t stride = (size_t)gridDim.x * blockDim.x;
    for (; i < total; i += stride) {
        int n = (int)(i >> 6);
        int h = (int)(i & 63);
        atomicAdd(&g[(size_t)batch[n] * 64 + h], (float)Xh[i]);
    }
}

// ---------------------------------------------------------------------------
// MLP head + log_softmax. One thread per graph (128 graphs; tiny).
// ---------------------------------------------------------------------------
__global__ void k_head(const float* __restrict__ g, const float* __restrict__ V0w,
                       const float* __restrict__ V0b, const float* __restrict__ V1w,
                       const float* __restrict__ V1b, float* __restrict__ out) {
    int t = blockIdx.x * blockDim.x + threadIdx.x;
    if (t >= NGRAPH) return;
    const float* gr = g + (size_t)t * 64;
    float h[64];
    for (int i = 0; i < 64; ++i) {
        float s = V0b[i];
        for (int j = 0; j < 64; ++j) s += gr[j] * V0w[i * 64 + j];
        h[i] = fmaxf(s, 0.f);
    }
    float lg[NCLASS];
    float mx = -1e30f;
    for (int c = 0; c < NCLASS; ++c) {
        float s = V1b[c];
        for (int i = 0; i < 64; ++i) s += h[i] * V1w[c * 64 + i];
        lg[c] = s;
        mx = fmaxf(mx, s);
    }
    float se = 0.f;
    for (int c = 0; c < NCLASS; ++c) se += expf(lg[c] - mx);
    float lse = mx + logf(se);
    for (int c = 0; c < NCLASS; ++c) out[t * NCLASS + c] = lg[c] - lse;
}

// ---------------------------------------------------------------------------
// Host orchestration.
// ---------------------------------------------------------------------------
extern "C" void kernel_launch(void* const* d_in, const int* in_sizes, int n_in,
                              void* d_out, int out_size, void* d_ws, size_t ws_size,
                              hipStream_t stream) {
    const float* features = (const float*)d_in[0];
    const float* ew  = (const float*)d_in[1];
    const float* W1  = (const float*)d_in[2];
    const float* O1  = (const float*)d_in[3];
    const float* W2  = (const float*)d_in[4];
    const float* O2  = (const float*)d_in[5];
    const float* W3  = (const float*)d_in[6];
    const float* O3  = (const float*)d_in[7];
    const float* V0w = (const float*)d_in[8];
    const float* V0b = (const float*)d_in[9];
    const float* V1w = (const float*)d_in[10];
    const float* V1b = (const float*)d_in[11];
    const int* eidx  = (const int*)d_in[12];
    const int* batch = (const int*)d_in[13];

    const int N = in_sizes[0] / NFEATC;   // 50000
    const int E = in_sizes[1];            // 800000
    const int* src = eidx;
    const int* dst = eidx + E;

    // workspace layout: f16 region first (all offsets 16B-aligned), then f32
    f16* hb = (f16*)d_ws;
    size_t off = 0;
    f16* Uth  = hb + off; off += (size_t)N * NFEATC;  // features^T f16 (N x 128)
    f16* bufA = hb + off; off += (size_t)N * 64;      // X ping
    f16* bufB = hb + off; off += (size_t)N * 64;      // X pong
    f16* Bth  = hb + off; off += (size_t)N * 64;      // b_Omega
    f16* Yth  = hb + off; off += (size_t)N * 64;      // X*A scratch / OmegaU
    f16* Wph  = hb + off; off += 64 * 64;             // projected W (f16)
    f16* Oh   = hb + off; off += 64 * NFEATC;         // Omega (f16)
    float* gpool = (float*)(hb + off);                // 128 x 64 f32

    // 1) transpose features -> node-major f16
    {
        dim3 tb(32, 8);
        dim3 tg((unsigned)((N + 31) / 32), (unsigned)((NFEATC + 31) / 32));
        k_transpose_h<<<tg, tb, 0, stream>>>(features, Uth, N, NFEATC);
    }

    const size_t nwords = (size_t)N * 32;             // N*64 f16 in u32 words
    const int gemmBlocks = ((N / 16) + 7) / 8;        // 8 waves/block, 1 tile each
    const int spmmBlocks = (int)(((size_t)E * 32 + 255) / 256);

    const float* Ws[3] = {W1, W2, W3};
    const float* Os[3] = {O1, O2, O3};
    const f16* Uin = Uth;
    int Kin = NFEATC;

    for (int l = 0; l < 3; ++l) {
        // Wp = proj_norm_inf(W, kappa)  (f16 out)
        k_proj<<<1, 64, 0, stream>>>(Ws[l], Wph, 64, KAPPA);
        // Omega -> f16
        k_cvt_h<<<(64 * Kin + 255) / 256, 256, 0, stream>>>(Os[l], Oh, 64 * Kin);
        // OU^T = Uin @ Omega^T -> Yth
        if (Kin == NFEATC)
            k_gemm_wmma<NFEATC, false, false><<<gemmBlocks, 256, 0, stream>>>(
                Uin, Oh, nullptr, Yth, N);
        else
            k_gemm_wmma<NHID, false, false><<<gemmBlocks, 256, 0, stream>>>(
                Uin, Oh, nullptr, Yth, N);
        // Bth = (Omega U) A  via packed-f16 edge scatter
        k_zero<<<1024, 256, 0, stream>>>((unsigned int*)Bth, nwords);
        k_spmm_h<<<spmmBlocks, 256, 0, stream>>>(Yth, src, dst, ew, Bth, E);
        // fixed-point iterations, X0 = Bth
        const f16* X = Bth;
        for (int it = 0; it < NITER; ++it) {
            k_zero<<<1024, 256, 0, stream>>>((unsigned int*)Yth, nwords);
            k_spmm_h<<<spmmBlocks, 256, 0, stream>>>(X, src, dst, ew, Yth, E);
            f16* Xn = (it & 1) ? bufB : bufA;
            k_gemm_wmma<NHID, true, true><<<gemmBlocks, 256, 0, stream>>>(
                Yth, Wph, Bth, Xn, N);
            X = Xn;
        }
        Uin = X;   // NITER=25 (odd count) -> ends in bufA
        Kin = NHID;
    }

    // pooling + head
    k_zero<<<64, 256, 0, stream>>>((unsigned int*)gpool, (size_t)NGRAPH * 64);
    const int poolBlocks = (int)(((size_t)N * 64 + 255) / 256);
    k_pool<<<poolBlocks, 256, 0, stream>>>(Uin, batch, gpool, N);
    k_head<<<1, 128, 0, stream>>>(gpool, V0w, V0b, V1w, V1b, (float*)d_out);
}
